// Speech_MSA_45432164057333
// MI455X (gfx1250) — matrix operand order repaired and checked by reading, not compiled, not run on hardware
//
#include <hip/hip_runtime.h>
#include <hip/hip_bf16.h>

typedef __bf16 bf16_t;
typedef __attribute__((ext_vector_type(16))) __bf16 v16bf;
typedef __attribute__((ext_vector_type(8)))  __bf16 v8bf;
typedef __attribute__((ext_vector_type(8)))  float  v8f;
typedef int v4i __attribute__((__vector_size__(16)));
typedef __attribute__((address_space(1))) v4i* as1_v4i_p;
typedef __attribute__((address_space(3))) v4i* as3_v4i_p;

#define EMBED   512
#define NHEADS  8
#define HDIM    64
#define WIN     17
#define PADW    8
#define BATCH   4
#define SEQ     2048
#define MROWS   (BATCH * SEQ)   /* 8192 */

// CDNA5 async global->LDS copy path (ASYNCcnt-tracked), with sync fallback.
#if defined(__gfx1250__) && __has_builtin(__builtin_amdgcn_global_load_async_to_lds_b128)
#define USE_ASYNC_LDS 1
#endif

__device__ __forceinline__ void cp16_to_lds(bf16_t* dst_lds, const bf16_t* src_g) {
#if USE_ASYNC_LDS
  __builtin_amdgcn_global_load_async_to_lds_b128(
      (as1_v4i_p)(src_g), (as3_v4i_p)(dst_lds), /*offset=*/0, /*cpol=*/0);
#else
  *reinterpret_cast<v8bf*>(dst_lds) = *reinterpret_cast<const v8bf*>(src_g);
#endif
}

__device__ __forceinline__ void wait_async0() {
#if USE_ASYNC_LDS
#if __has_builtin(__builtin_amdgcn_s_wait_asynccnt)
  __builtin_amdgcn_s_wait_asynccnt(0);
#else
  asm volatile("s_wait_asynccnt 0" ::: "memory");
#endif
#endif
}

__device__ __forceinline__ bf16_t f2bf(float f) {
  // round-to-nearest-even f32 -> bf16
  unsigned u = __builtin_bit_cast(unsigned, f);
  u = u + 0x7FFFu + ((u >> 16) & 1u);
  unsigned short h = (unsigned short)(u >> 16);
  return __builtin_bit_cast(bf16_t, h);
}

// ---------------------------------------------------------------------------
// f32 -> bf16 elementwise convert (vectorized x4)
// ---------------------------------------------------------------------------
__global__ __launch_bounds__(256) void cvt_f32_bf16_x4(
    const float* __restrict__ in, bf16_t* __restrict__ out, int n4) {
  int i = blockIdx.x * blockDim.x + threadIdx.x;
  if (i < n4) {
    float4 v = reinterpret_cast<const float4*>(in)[i];
    bf16_t* o = out + (size_t)i * 4;
    o[0] = f2bf(v.x); o[1] = f2bf(v.y); o[2] = f2bf(v.z); o[3] = f2bf(v.w);
  }
}

// ---------------------------------------------------------------------------
// transpose + convert: in (K x N) f32 row-major -> out (N x K) bf16 row-major
// ---------------------------------------------------------------------------
__global__ __launch_bounds__(256) void transpose_cvt(
    const float* __restrict__ in, bf16_t* __restrict__ out, int K, int N) {
  long idx = (long)blockIdx.x * blockDim.x + threadIdx.x;
  if (idx < (long)K * N) {
    int n = (int)(idx / K);
    int k = (int)(idx % K);
    out[idx] = f2bf(in[(size_t)k * N + n]);
  }
}

// ---------------------------------------------------------------------------
// bf16 WMMA GEMM: C[M,N](f32) = A[M,K](bf16) @ Bt[N,K](bf16)^T + bias[N]
// block = 128 threads (4 waves), tile 128x64, K-step 32.
// Each wave: 32 rows x 64 cols = 8 WMMA accumulators per K-step.
// A tile double-buffered in LDS via GLOBAL_LOAD_ASYNC_TO_LDS_B128.
// B fragments register-double-buffered (2x unrolled K loop, no reg copies).
// ---------------------------------------------------------------------------
#define BM 128
#define BN 64
#define BK 32
#define LDA 40   /* padded halves per LDS A row (80 bytes, keeps 16B alignment) */

__global__ __launch_bounds__(128) void gemm_bf16_wmma(
    const bf16_t* __restrict__ A, const bf16_t* __restrict__ Bt,
    const float* __restrict__ bias, float* __restrict__ C,
    int M, int N, int K) {
  __shared__ bf16_t sA[2][BM * LDA];

  const int tid  = threadIdx.x;
  const int lane = tid & 31;
  const int wave = tid >> 5;        // 0..3
  const int m0 = blockIdx.y * BM;
  const int n0 = blockIdx.x * BN;
  const int KT = K / BK;            // 16 (always even here)

  // Stage one 128x32 bf16 A tile (8 KB): 512 16B chunks, 4 per thread.
  auto stage = [&](int buf, int k0) {
#pragma unroll
    for (int i = 0; i < 4; ++i) {
      int c  = tid + i * 128;       // chunk id 0..511
      int r  = c >> 2;              // tile row 0..127
      int kc = (c & 3) * 8;         // k-offset {0,8,16,24}
      cp16_to_lds(&sA[buf][r * LDA + kc],
                  A + (size_t)(m0 + r) * K + k0 + kc);
    }
  };

  // B fragment (contiguous k order, same as A): lane<16 -> col n=lane, K=0..15;
  // lane>=16 -> col n=lane-16, K=16..31. Bt is N x K so this is contiguous.
  auto loadB = [&](int k0, v16bf* frag) {
    const int nn = lane & 15;
    const int kk = (lane >> 4) * 16;
#pragma unroll
    for (int j = 0; j < 4; ++j)
      frag[j] = *reinterpret_cast<const v16bf*>(
          Bt + (size_t)(n0 + j * 16 + nn) * K + k0 + kk);
  };

  // 16-bit A 16x32 layout (matches B): lane<16 holds row (lane&15), K 0..15;
  // lane>=16 holds K 16..31.
  const int ar0 = wave * 32 + (lane & 15);
  const int kh  = (lane >> 4) * 16;
  auto loadA = [&](int buf, int row) -> v16bf {
    v8bf lo = *reinterpret_cast<const v8bf*>(&sA[buf][row * LDA + kh]);
    v8bf hi = *reinterpret_cast<const v8bf*>(&sA[buf][row * LDA + kh + 8]);
    return __builtin_shufflevector(lo, hi, 0, 1, 2, 3, 4, 5, 6, 7,
                                   8, 9, 10, 11, 12, 13, 14, 15);
  };

  v8f acc0[4], acc1[4];
#pragma unroll
  for (int j = 0; j < 4; ++j)
#pragma unroll
    for (int e = 0; e < 8; ++e) { acc0[j][e] = 0.0f; acc1[j][e] = 0.0f; }

  auto compute = [&](int buf, const v16bf* bf) {
    v16bf a0 = loadA(buf, ar0);
    v16bf a1 = loadA(buf, ar0 + 16);
#pragma unroll
    for (int j = 0; j < 4; ++j)
      acc0[j] = __builtin_amdgcn_wmma_f32_16x16x32_bf16(
          false, a0, false, bf[j], (short)0, acc0[j], false, false);
#pragma unroll
    for (int j = 0; j < 4; ++j)
      acc1[j] = __builtin_amdgcn_wmma_f32_16x16x32_bf16(
          false, a1, false, bf[j], (short)0, acc1[j], false, false);
  };

  v16bf bA[4], bB[4];
  stage(0, 0);
  loadB(0, bA);
  wait_async0();
  __syncthreads();

  for (int kt = 0; kt < KT; kt += 2) {
    // even step: consume buf0/bA, prefetch (kt+1) -> buf1/bB
    if (kt + 1 < KT) {
      stage(1, (kt + 1) * BK);
      loadB((kt + 1) * BK, bB);
      if (kt + 2 < KT)
        __builtin_prefetch(A + (size_t)(m0 + tid) * K + (kt + 2) * BK, 0, 3);
    }
    compute(0, bA);
    wait_async0();
    __syncthreads();

    // odd step: consume buf1/bB, prefetch (kt+2) -> buf0/bA
    if (kt + 2 < KT) {
      stage(0, (kt + 2) * BK);
      loadB((kt + 2) * BK, bA);
    }
    compute(1, bB);
    wait_async0();
    __syncthreads();
  }

  // C/D f32 16x16 layout: VGPR i -> row i (lanes 0..15) / row 8+i (lanes 16..31)
  const int cn = lane & 15;
  const int rb = m0 + wave * 32 + (lane >> 4) * 8;
#pragma unroll
  for (int j = 0; j < 4; ++j) {
    int col = n0 + j * 16 + cn;
    float bv = bias[col];
#pragma unroll
    for (int i = 0; i < 8; ++i)
      C[(size_t)(rb + i) * N + col] = acc0[j][i] + bv;
#pragma unroll
    for (int i = 0; i < 8; ++i)
      C[(size_t)(rb + 16 + i) * N + col] = acc1[j][i] + bv;
  }
}

// ---------------------------------------------------------------------------
// Windowed local attention: one wave32 per (b, h, t).
// qkv: (MROWS, 3E) f32; ctx: (MROWS, E) bf16 with head layout h*HDIM+d.
// Matches reference semantics: zero-padded keys score 0 and participate in
// softmax; padded values contribute 0 to the output.
// ---------------------------------------------------------------------------
__global__ __launch_bounds__(256) void local_attn(
    const float* __restrict__ qkv, bf16_t* __restrict__ ctx) {
  const int lane = threadIdx.x & 31;
  const int widx = blockIdx.x * 8 + (threadIdx.x >> 5);
  const int t = widx & (SEQ - 1);
  const int h = (widx >> 11) & (NHEADS - 1);
  const int b = widx >> 14;
  const float scaling = 0.125f;  // HDIM^-0.5

  const int qc = h * HDIM;
  const int kc = EMBED + h * HDIM;
  const int vc = 2 * EMBED + h * HDIM;
  const size_t row = (size_t)(b * SEQ + t);

  const float q0 = qkv[row * (3 * EMBED) + qc + lane] * scaling;
  const float q1 = qkv[row * (3 * EMBED) + qc + lane + 32] * scaling;

  float s[WIN];
#pragma unroll
  for (int w = 0; w < WIN; ++w) {
    int kt = t - PADW + w;
    float p = 0.0f;
    if (kt >= 0 && kt < SEQ) {
      const float* kp = qkv + (size_t)(b * SEQ + kt) * (3 * EMBED) + kc;
      p = q0 * kp[lane] + q1 * kp[lane + 32];
    }
#pragma unroll
    for (int off = 16; off > 0; off >>= 1) p += __shfl_xor(p, off, 32);
    s[w] = p;  // zero-padded keys => score 0, included in softmax
  }

  float m = s[0];
#pragma unroll
  for (int w = 1; w < WIN; ++w) m = fmaxf(m, s[w]);
  float sum = 0.0f;
#pragma unroll
  for (int w = 0; w < WIN; ++w) { s[w] = __expf(s[w] - m); sum += s[w]; }
  const float inv = 1.0f / sum;

  float o0 = 0.0f, o1 = 0.0f;
#pragma unroll
  for (int w = 0; w < WIN; ++w) {
    int kt = t - PADW + w;
    if (kt >= 0 && kt < SEQ) {
      const float* vp = qkv + (size_t)(b * SEQ + kt) * (3 * EMBED) + vc;
      float pw = s[w] * inv;
      o0 += pw * vp[lane];
      o1 += pw * vp[lane + 32];
    }
  }

  bf16_t* op = ctx + row * EMBED + h * HDIM;
  op[lane] = f2bf(o0);
  op[lane + 32] = f2bf(o1);
}

// ---------------------------------------------------------------------------
extern "C" void kernel_launch(void* const* d_in, const int* in_sizes, int n_in,
                              void* d_out, int out_size, void* d_ws, size_t ws_size,
                              hipStream_t stream) {
  const float* x     = (const float*)d_in[0];
  const float* w_qkv = (const float*)d_in[1];
  const float* b_qkv = (const float*)d_in[2];
  const float* w_out = (const float*)d_in[3];
  const float* b_out = (const float*)d_in[4];
  float* out = (float*)d_out;
  (void)in_sizes; (void)n_in; (void)out_size; (void)ws_size;

  char* ws = (char*)d_ws;
  size_t off = 0;
  auto alloc = [&](size_t bytes) -> void* {
    void* p = ws + off;
    off = (off + bytes + 255) & ~(size_t)255;
    return p;
  };
  bf16_t* x_bf   = (bf16_t*)alloc((size_t)MROWS * EMBED * 2);        // 8.39 MB
  bf16_t* wqkv_t = (bf16_t*)alloc((size_t)3 * EMBED * EMBED * 2);    // 1.57 MB
  bf16_t* wout_t = (bf16_t*)alloc((size_t)EMBED * EMBED * 2);        // 0.52 MB
  float*  qkv    = (float*) alloc((size_t)MROWS * 3 * EMBED * 4);    // 50.3 MB
  bf16_t* ctx    = (bf16_t*)alloc((size_t)MROWS * EMBED * 2);        // 8.39 MB

  // 1) x -> bf16
  {
    int n4 = MROWS * EMBED / 4;
    cvt_f32_bf16_x4<<<n4 / 256, 256, 0, stream>>>(x, x_bf, n4);
  }
  // 2) weights -> transposed bf16 (N x K)
  transpose_cvt<<<(512 * 1536) / 256, 256, 0, stream>>>(w_qkv, wqkv_t, 512, 1536);
  transpose_cvt<<<(512 * 512) / 256, 256, 0, stream>>>(w_out, wout_t, 512, 512);

  // 3) qkv = x @ w_qkv + b_qkv   (M=8192, N=1536, K=512)
  gemm_bf16_wmma<<<dim3(1536 / BN, MROWS / BM), 128, 0, stream>>>(
      x_bf, wqkv_t, b_qkv, qkv, MROWS, 1536, 512);

  // 4) windowed attention -> ctx (bf16)
  local_attn<<<(BATCH * NHEADS * SEQ) / 8, 256, 0, stream>>>(qkv, ctx);

  // 5) out = ctx @ w_out + b_out  (M=8192, N=512, K=512)
  gemm_bf16_wmma<<<dim3(512 / BN, MROWS / BM), 128, 0, stream>>>(
      ctx, wout_t, b_out, out, MROWS, 512, 512);
}
